// T2T_67654324847048
// MI455X (gfx1250) — matrix operand order repaired and checked by reading, not compiled, not run on hardware
//
#include <hip/hip_runtime.h>
#include <hip/hip_bf16.h>
#include <math.h>

// ---------------------------------------------------------------------------
// MI455X (gfx1250). All GEMMs via v_wmma_f32_16x16x32_bf16 (fp32 accumulate).
// v2 changes:
//  - vocab GEMM: 64 rows per block, Wout tiles staged in LDS and shared by 4
//    row-tile waves (4x less L2 traffic, 64 FLOP per B-byte).
//  - stack LSTM: x-projection for all 321 steps hoisted into one batched WMMA
//    GEMM (XZ); the persistent workgroup only runs the sequential h @ Whh^T.
// ---------------------------------------------------------------------------

typedef __bf16 bf16_t;
typedef __attribute__((ext_vector_type(16))) __bf16 v16bf;
typedef __attribute__((ext_vector_type(8)))  __bf16 v8bf;
typedef __attribute__((ext_vector_type(8)))  float   v8f;

#define SIZEH  512
#define VOCAB  32000
#define T_TOK  256
#define BATCH  16
#define KCH    8
#define GRP    32                  // T/K
#define GB     512                 // GRP*BATCH
#define H4     2048                // 4*SIZEH
#define NSTEP  321                 // 1 + GRP*(KCH+2)
#define NROWS  (T_TOK*BATCH)       // 4096
#define VTILES (VOCAB/16)          // 2000
#define NPART  16                  // partial-logsumexp slots (8 blocks x 2)

// ---- WMMA fragment helpers (layouts per CDNA5 ISA 7.12.2) ------------------

// A fragment: 16x32 bf16 tile, A pre-offset to row m0, row-major lda (elems).
// lane<16: row=lane, K = {k0..k0+7} U {k0+16..k0+23}
// lane>=16: row=lane-16, K = {k0+8..k0+15} U {k0+24..k0+31}
__device__ __forceinline__ v16bf load_a_frag(const bf16_t* __restrict__ A,
                                             int lda, int k0, int lane) {
  int half = lane >> 4, r = lane & 15;
  const bf16_t* p = A + (size_t)r * lda + k0 + half * 8;
  v8bf lo = *(const v8bf*)p;
  v8bf hi = *(const v8bf*)(p + 16);
  return __builtin_shufflevector(lo, hi, 0,1,2,3,4,5,6,7,8,9,10,11,12,13,14,15);
}

// B fragment: 32x16, element B[k][n] = W[n][k] with W row-major [N,K].
// lane<16: col=lane, K = {k0..k0+7} U {k0+16..k0+23}
// lane>=16: col=lane-16, K = {k0+8..k0+15} U {k0+24..k0+31}
__device__ __forceinline__ v16bf load_b_frag(const bf16_t* __restrict__ W,
                                             int ldw, int n0, int k0, int lane) {
  int half = lane >> 4, c = lane & 15;
  const bf16_t* p = W + (size_t)(n0 + c) * ldw + k0 + half * 8;
  v8bf lo = *(const v8bf*)p;
  v8bf hi = *(const v8bf*)(p + 16);
  return __builtin_shufflevector(lo, hi, 0,1,2,3,4,5,6,7,8,9,10,11,12,13,14,15);
}

__device__ __forceinline__ v8f wmma_bf16(v16bf a, v16bf b, v8f c) {
  return __builtin_amdgcn_wmma_f32_16x16x32_bf16(false, a, false, b,
                                                 (short)0, c, false, false);
}

__device__ __forceinline__ float sigm(float x) { return 1.f / (1.f + __expf(-x)); }

// ---- small utility kernels -------------------------------------------------

__global__ void k_f32_to_bf16(const float* __restrict__ src,
                              bf16_t* __restrict__ dst, int n) {
  int i = blockIdx.x * 256 + threadIdx.x;
  if (i < n) dst[i] = (bf16_t)src[i];
}

__global__ void k_embed(const float* __restrict__ emb, const int* __restrict__ tokens,
                        bf16_t* __restrict__ E) {
  int idx = blockIdx.x * 256 + threadIdx.x;      // NROWS*SIZEH
  int row = idx >> 9, s = idx & 511;
  E[idx] = (bf16_t)emb[(size_t)tokens[row] * SIZEH + s];
}

// EG[(k*GB + g*16 + b), s] = E[((g*8+k)*16 + b), s]
__global__ void k_egperm(const bf16_t* __restrict__ E, bf16_t* __restrict__ EG) {
  int idx = blockIdx.x * 256 + threadIdx.x;      // NROWS*SIZEH
  int row = idx >> 9, s = idx & 511;
  int k = row >> 9, gb = row & 511;
  int g = gb >> 4, b = gb & 15;
  int src = (g * KCH + k) * BATCH + b;
  EG[idx] = E[((size_t)src << 9) + s];
}

// Stack-LSTM input sequence (except compose rows filled later).
__global__ void k_fill_xs(const float* __restrict__ emb, const int* __restrict__ tokens,
                          const bf16_t* __restrict__ E, bf16_t* __restrict__ XS) {
  int j = blockIdx.x;                            // 0..NSTEP-1
  bool isOpen;
  int erow_base = 0;
  if (j == 0) { isOpen = true; }
  else {
    int r = (j - 1) % 10, g = (j - 1) / 10;
    if (r == 0) isOpen = true;
    else if (r <= 8) { isOpen = false; erow_base = (g * KCH + (r - 1)) * BATCH; }
    else return;                                 // compose row: written later
  }
  int tok0 = tokens[0];
  for (int i = threadIdx.x; i < 16 * SIZEH; i += 256) {
    int b = i >> 9, s = i & 511;
    bf16_t v = isOpen ? (bf16_t)emb[(size_t)tok0 * SIZEH + s]
                      : E[((size_t)(erow_base + b) << 9) + s];
    XS[(((size_t)j * 16) << 9) + i] = v;
  }
}

// Elementwise LSTM cell over GB rows (gate order i,f,g,o in Z[m, 4*SIZEH]).
__global__ void k_lstm_cell(const float* __restrict__ Z, float* __restrict__ C,
                            bf16_t* __restrict__ H) {
  int idx = blockIdx.x * 256 + threadIdx.x;      // GB*SIZEH
  int m = idx >> 9, s = idx & 511;
  const float* z = Z + (size_t)m * H4;
  float iv = sigm(z[s]);
  float fv = sigm(z[SIZEH + s]);
  float gv = tanhf(z[2 * SIZEH + s]);
  float ov = sigm(z[3 * SIZEH + s]);
  float c = fv * C[idx] + iv * gv;
  C[idx] = c;
  H[idx] = (bf16_t)(ov * tanhf(c));
}

__global__ void k_hcat(const bf16_t* __restrict__ hf, const bf16_t* __restrict__ hb,
                       bf16_t* __restrict__ HC) {
  int idx = blockIdx.x * 256 + threadIdx.x;      // GB * 2*SIZEH
  int m = idx >> 10, c = idx & 1023;
  HC[idx] = (c < SIZEH) ? hf[(m << 9) + c] : hb[(m << 9) + c - SIZEH];
}

// comp = tanh(ZC), scattered into XS at step j = 10*(g+1).
__global__ void k_comp_scatter(const float* __restrict__ ZC, bf16_t* __restrict__ XS) {
  int idx = blockIdx.x * 256 + threadIdx.x;      // GB*SIZEH
  int m = idx >> 9, s = idx & 511;
  int g = m >> 4, b = m & 15;
  int j = 10 + 10 * g;
  XS[(((size_t)j * 16 + b) << 9) + s] = (bf16_t)tanhf(ZC[idx]);
}

// ---- generic dual-operand WMMA GEMM: Z = A1*W1^T (+ A2*W2^T) + b1 (+ b2) ---
__global__ __launch_bounds__(256)
void k_gemm_dual(const bf16_t* __restrict__ A1, int lda1,
                 const bf16_t* __restrict__ W1, int K1,
                 const bf16_t* __restrict__ A2, int lda2,
                 const bf16_t* __restrict__ W2, int K2,
                 const float* __restrict__ b1, const float* __restrict__ b2,
                 float* __restrict__ Z, int N) {
  int wave = threadIdx.x >> 5, lane = threadIdx.x & 31;
  int n0 = (blockIdx.x * 8 + wave) * 16;
  if (n0 >= N) return;
  int m0 = blockIdx.y * 16;
  v8f acc = {};
  const bf16_t* A1t = A1 + (size_t)m0 * lda1;
  for (int k0 = 0; k0 < K1; k0 += 32)
    acc = wmma_bf16(load_a_frag(A1t, lda1, k0, lane),
                    load_b_frag(W1, K1, n0, k0, lane), acc);
  if (A2) {
    const bf16_t* A2t = A2 + (size_t)m0 * lda2;
    for (int k0 = 0; k0 < K2; k0 += 32)
      acc = wmma_bf16(load_a_frag(A2t, lda2, k0, lane),
                      load_b_frag(W2, K2, n0, k0, lane), acc);
  }
  int half = lane >> 4, nc = lane & 15, mb = half * 8;
  float bias = 0.f;
  if (b1) bias += b1[n0 + nc];
  if (b2) bias += b2[n0 + nc];
  float* zr = Z + (size_t)(m0 + mb) * N + n0 + nc;
#pragma unroll
  for (int v = 0; v < 8; ++v) zr[(size_t)v * N] = acc[v] + bias;
}

// ---- persistent stack LSTM: one workgroup, 32 waves, 321 sequential steps --
// x-projection (incl. both biases) is precomputed in XZ[NSTEP*16, H4]; only
// the sequential h @ Whh^T runs here. Wave w owns hidden columns s0=w*16 and
// computes its four gate tiles, so the cell update is in-register; h (bf16)
// lives in LDS and feeds WMMA A-fragments the next step.
__global__ __launch_bounds__(1024)
void k_stack_lstm(const float* __restrict__ XZ,
                  const bf16_t* __restrict__ Whh,
                  bf16_t* __restrict__ HP) {
  __shared__ __align__(64) bf16_t h_sh[16 * SIZEH];
  int tid = threadIdx.x, wave = tid >> 5, lane = tid & 31;
  int half = lane >> 4, nc = lane & 15, mb = half * 8;
  int s0 = wave * 16;
  for (int i = tid; i < 16 * SIZEH; i += 1024) h_sh[i] = (bf16_t)0.f;
  v8f c = {};
  __syncthreads();
  for (int j = 0; j < NSTEP; ++j) {
    v8f zi, zf, zg, zo;
#pragma unroll
    for (int v = 0; v < 8; ++v) {                     // init from precomputed x-part
      const float* zr = XZ + (size_t)(j * 16 + mb + v) * H4 + s0 + nc;
      zi[v] = zr[0];
      zf[v] = zr[SIZEH];
      zg[v] = zr[2 * SIZEH];
      zo[v] = zr[3 * SIZEH];
    }
    for (int k0 = 0; k0 < SIZEH; k0 += 32) {          // h @ Whh^T (h in LDS)
      v16bf a = load_a_frag(h_sh, SIZEH, k0, lane);
      zi = wmma_bf16(a, load_b_frag(Whh, SIZEH, s0,             k0, lane), zi);
      zf = wmma_bf16(a, load_b_frag(Whh, SIZEH, SIZEH + s0,     k0, lane), zf);
      zg = wmma_bf16(a, load_b_frag(Whh, SIZEH, 2 * SIZEH + s0, k0, lane), zg);
      zo = wmma_bf16(a, load_b_frag(Whh, SIZEH, 3 * SIZEH + s0, k0, lane), zo);
    }
    float hval[8];
#pragma unroll
    for (int v = 0; v < 8; ++v) {
      float cn = sigm(zf[v]) * c[v] + sigm(zi[v]) * tanhf(zg[v]);
      c[v] = cn;
      hval[v] = sigm(zo[v]) * tanhf(cn);
    }
    __syncthreads();                                  // everyone done reading h_sh
#pragma unroll
    for (int v = 0; v < 8; ++v)
      h_sh[(mb + v) * SIZEH + s0 + nc] = (bf16_t)hval[v];
    int jj = j - 1;                                   // hs[j] used iff pos==j
    if (jj >= 0) {
      int g = jj / 10, r = jj - g * 10;
      if (r < 8) {
        int t = g * KCH + r;
#pragma unroll
        for (int v = 0; v < 8; ++v)
          HP[((size_t)(t * 16 + mb + v)) * SIZEH + s0 + nc] = (bf16_t)hval[v];
      }
    }
    __syncthreads();                                  // h_sh ready for next step
  }
}

// ---- fused vocab GEMM + partial online logsumexp ---------------------------
// grid = (8, 64): blockIdx.y = 64-row block; 8 waves = 4 row tiles x 2 column
// slots. Each column slot stages its 16x512 Wout tile in LDS once and all 4
// row-tile waves consume it (4x operand reuse -> 64 FLOP per global B-byte).
__global__ __launch_bounds__(256)
void k_logits_lse(const bf16_t* __restrict__ HP, const bf16_t* __restrict__ WO,
                  const float* __restrict__ bout, const int* __restrict__ tokens,
                  float* __restrict__ pmax, float* __restrict__ psum,
                  float* __restrict__ ptv) {
  __shared__ __align__(64) bf16_t btile[2][16 * SIZEH];   // 2 x 16KB
  int tid = threadIdx.x;
  int wave = tid >> 5, lane = tid & 31;
  int half = lane >> 4, nc = lane & 15, mb = half * 8;
  int rtl = wave & 3;                       // local row tile 0..3
  int cslot = wave >> 2;                    // 0..1
  int m0 = blockIdx.y * 64 + rtl * 16;
  int slot = blockIdx.x * 2 + cslot;        // 0..NPART-1
  int lt = tid & 127;                       // thread id within column slot
  int frow = lt >> 3, fseg = lt & 7;        // cooperative-fill mapping

  const bf16_t* At = HP + (size_t)m0 * SIZEH;
  v16bf af[16];
#pragma unroll
  for (int kt = 0; kt < 16; ++kt) af[kt] = load_a_frag(At, SIZEH, kt * 32, lane);
  int tok[8];
#pragma unroll
  for (int v = 0; v < 8; ++v) tok[v] = tokens[m0 + mb + v];
  float rmax[8], rsum[8], tv[8];
#pragma unroll
  for (int v = 0; v < 8; ++v) { rmax[v] = -1e30f; rsum[v] = 0.f; tv[v] = -1e30f; }

  for (int nt = slot; nt < VTILES; nt += NPART) {     // 125 iters, all waves
    int n0 = nt * 16;
    __syncthreads();                                  // previous tile consumed
    {
      const bf16_t* src = WO + (size_t)(n0 + frow) * SIZEH + fseg * 64;
      bf16_t* dst = &btile[cslot][frow * SIZEH + fseg * 64];
      __builtin_prefetch(src + (size_t)NPART * 16 * SIZEH, 0, 0);
#pragma unroll
      for (int q = 0; q < 4; ++q)
        *(v16bf*)(dst + q * 16) = *(const v16bf*)(src + q * 16);
    }
    __syncthreads();                                  // tile staged
    v8f acc = {};
#pragma unroll
    for (int kt = 0; kt < 16; ++kt)
      acc = wmma_bf16(af[kt], load_b_frag(btile[cslot], SIZEH, 0, kt * 32, lane), acc);
    int col = n0 + nc;
    float bias = bout[col];
#pragma unroll
    for (int v = 0; v < 8; ++v) {
      float val = acc[v] + bias;
      float m = rmax[v];
      if (val > m) { rsum[v] = rsum[v] * __expf(m - val) + 1.f; rmax[v] = val; }
      else         { rsum[v] += __expf(val - m); }
      if (col == tok[v]) tv[v] = val;
    }
  }
  // merge the 16 lanes that share each row (xor masks stay within half-wave)
#pragma unroll
  for (int mask = 1; mask < 16; mask <<= 1) {
#pragma unroll
    for (int v = 0; v < 8; ++v) {
      float omx = __shfl_xor(rmax[v], mask, 32);
      float osm = __shfl_xor(rsum[v], mask, 32);
      float nmx = fmaxf(rmax[v], omx);
      rsum[v] = rsum[v] * __expf(rmax[v] - nmx) + osm * __expf(omx - nmx);
      rmax[v] = nmx;
      float otv = __shfl_xor(tv[v], mask, 32);
      tv[v] = fmaxf(tv[v], otv);
    }
  }
  if (nc == 0) {
#pragma unroll
    for (int v = 0; v < 8; ++v) {
      int m = m0 + mb + v;
      pmax[m * NPART + slot] = rmax[v];
      psum[m * NPART + slot] = rsum[v];
      ptv [m * NPART + slot] = tv[v];
    }
  }
}

__global__ void k_combine(const float* __restrict__ pmax, const float* __restrict__ psum,
                          const float* __restrict__ ptv, float* __restrict__ out) {
  int m = blockIdx.x * 256 + threadIdx.x;
  if (m >= NROWS) return;
  float mx = -1e30f, sm = 0.f, tv = -1e30f;
  for (int p = 0; p < NPART; ++p) {
    float omx = pmax[m * NPART + p], osm = psum[m * NPART + p];
    float nmx = fmaxf(mx, omx);
    sm = sm * __expf(mx - nmx) + osm * __expf(omx - nmx);
    mx = nmx;
    tv = fmaxf(tv, ptv[m * NPART + p]);
  }
  out[m] = (__logf(sm) + mx) - tv;   // -log_softmax at target
}

// ---------------------------------------------------------------------------
extern "C" void kernel_launch(void* const* d_in, const int* in_sizes, int n_in,
                              void* d_out, int out_size, void* d_ws, size_t ws_size,
                              hipStream_t stream) {
  (void)in_sizes; (void)n_in; (void)out_size; (void)ws_size;
  const int*   tokens = (const int*)  d_in[0];
  const float* emb    = (const float*)d_in[1];
  const float* Wih_s  = (const float*)d_in[2];
  const float* Whh_s  = (const float*)d_in[3];
  const float* bih_s  = (const float*)d_in[4];
  const float* bhh_s  = (const float*)d_in[5];
  const float* Wih_f  = (const float*)d_in[6];
  const float* Whh_f  = (const float*)d_in[7];
  const float* bih_f  = (const float*)d_in[8];
  const float* bhh_f  = (const float*)d_in[9];
  const float* Wih_b  = (const float*)d_in[10];
  const float* Whh_b  = (const float*)d_in[11];
  const float* bih_b  = (const float*)d_in[12];
  const float* bhh_b  = (const float*)d_in[13];
  const float* Wc     = (const float*)d_in[14];
  const float* bc     = (const float*)d_in[15];
  const float* Wout   = (const float*)d_in[16];
  const float* bout   = (const float*)d_in[17];
  float* out = (float*)d_out;

  char* ws = (char*)d_ws;
  size_t off = 0;
  auto alloc = [&](size_t bytes) -> void* {
    void* p = ws + off;
    off = (off + bytes + 255) & ~(size_t)255;
    return p;
  };
  bf16_t* WihS = (bf16_t*)alloc((size_t)H4 * SIZEH * 2);
  bf16_t* WhhS = (bf16_t*)alloc((size_t)H4 * SIZEH * 2);
  bf16_t* WihF = (bf16_t*)alloc((size_t)H4 * SIZEH * 2);
  bf16_t* WhhF = (bf16_t*)alloc((size_t)H4 * SIZEH * 2);
  bf16_t* WihB = (bf16_t*)alloc((size_t)H4 * SIZEH * 2);
  bf16_t* WhhB = (bf16_t*)alloc((size_t)H4 * SIZEH * 2);
  bf16_t* WcB  = (bf16_t*)alloc((size_t)SIZEH * 2 * SIZEH * 2);
  bf16_t* WoB  = (bf16_t*)alloc((size_t)VOCAB * SIZEH * 2);
  bf16_t* E    = (bf16_t*)alloc((size_t)NROWS * SIZEH * 2);
  bf16_t* EG   = (bf16_t*)alloc((size_t)NROWS * SIZEH * 2);
  bf16_t* XS   = (bf16_t*)alloc((size_t)NSTEP * 16 * SIZEH * 2);
  float*  XZ   = (float*) alloc((size_t)NSTEP * 16 * H4 * 4);
  float*  Zcmp = (float*) alloc((size_t)GB * H4 * 4);
  float*  Cf   = (float*) alloc((size_t)GB * SIZEH * 4);
  float*  Cb   = (float*) alloc((size_t)GB * SIZEH * 4);
  bf16_t* Hf   = (bf16_t*)alloc((size_t)GB * SIZEH * 2);
  bf16_t* Hb   = (bf16_t*)alloc((size_t)GB * SIZEH * 2);
  bf16_t* HC   = (bf16_t*)alloc((size_t)GB * 2 * SIZEH * 2);
  float*  ZC2  = (float*) alloc((size_t)GB * SIZEH * 4);
  bf16_t* HP   = (bf16_t*)alloc((size_t)NROWS * SIZEH * 2);
  float*  Pmax = (float*) alloc((size_t)NROWS * NPART * 4);
  float*  Psum = (float*) alloc((size_t)NROWS * NPART * 4);
  float*  Ptv  = (float*) alloc((size_t)NROWS * NPART * 4);

  auto cvt = [&](const float* s, bf16_t* d, int n) {
    k_f32_to_bf16<<<(n + 255) / 256, 256, 0, stream>>>(s, d, n);
  };
  cvt(Wih_s, WihS, H4 * SIZEH);
  cvt(Whh_s, WhhS, H4 * SIZEH);
  cvt(Wih_f, WihF, H4 * SIZEH);
  cvt(Whh_f, WhhF, H4 * SIZEH);
  cvt(Wih_b, WihB, H4 * SIZEH);
  cvt(Whh_b, WhhB, H4 * SIZEH);
  cvt(Wc,    WcB,  SIZEH * 2 * SIZEH);
  cvt(Wout,  WoB,  VOCAB * SIZEH);

  k_embed <<<NROWS * SIZEH / 256, 256, 0, stream>>>(emb, tokens, E);
  k_egperm<<<NROWS * SIZEH / 256, 256, 0, stream>>>(E, EG);
  k_fill_xs<<<NSTEP, 256, 0, stream>>>(emb, tokens, E, XS);

  hipMemsetAsync(Cf, 0, (size_t)GB * SIZEH * 4, stream);
  hipMemsetAsync(Cb, 0, (size_t)GB * SIZEH * 4, stream);
  hipMemsetAsync(Hf, 0, (size_t)GB * SIZEH * 2, stream);
  hipMemsetAsync(Hb, 0, (size_t)GB * SIZEH * 2, stream);

  dim3 gC(H4 / 16 / 8, GB / 16);                 // (16, 32)
  // compose forward direction
  for (int k = 0; k < KCH; ++k) {
    k_gemm_dual<<<gC, 256, 0, stream>>>(EG + (size_t)k * GB * SIZEH, SIZEH, WihF, SIZEH,
                                        Hf, SIZEH, WhhF, SIZEH,
                                        bih_f, bhh_f, Zcmp, H4);
    k_lstm_cell<<<GB * SIZEH / 256, 256, 0, stream>>>(Zcmp, Cf, Hf);
  }
  // compose backward direction (reversed child order)
  for (int k = 0; k < KCH; ++k) {
    k_gemm_dual<<<gC, 256, 0, stream>>>(EG + (size_t)(KCH - 1 - k) * GB * SIZEH, SIZEH,
                                        WihB, SIZEH, Hb, SIZEH, WhhB, SIZEH,
                                        bih_b, bhh_b, Zcmp, H4);
    k_lstm_cell<<<GB * SIZEH / 256, 256, 0, stream>>>(Zcmp, Cb, Hb);
  }

  k_hcat<<<GB * 2 * SIZEH / 256, 256, 0, stream>>>(Hf, Hb, HC);
  dim3 gP(SIZEH / 16 / 8, GB / 16);              // (4, 32)
  k_gemm_dual<<<gP, 256, 0, stream>>>(HC, 2 * SIZEH, WcB, 2 * SIZEH,
                                      (const bf16_t*)nullptr, 0,
                                      (const bf16_t*)nullptr, 0,
                                      bc, (const float*)nullptr, ZC2, SIZEH);
  k_comp_scatter<<<GB * SIZEH / 256, 256, 0, stream>>>(ZC2, XS);

  // batched x-projection for the stack LSTM: XZ = XS @ Wih_s^T + bih + bhh
  k_gemm_dual<<<dim3(H4 / 16 / 8, NSTEP), 256, 0, stream>>>(
      XS, SIZEH, WihS, SIZEH,
      (const bf16_t*)nullptr, 0, (const bf16_t*)nullptr, 0,
      bih_s, bhh_s, XZ, H4);

  // persistent sequential stack LSTM (single workgroup, 32 waves)
  k_stack_lstm<<<1, 1024, 0, stream>>>(XZ, WhhS, HP);

  // fused vocab projection + partial logsumexp, then combine into loss
  k_logits_lse<<<dim3(NPART / 2, NROWS / 64), 256, 0, stream>>>(
      HP, WoB, bout, tokens, Pmax, Psum, Ptv);
  k_combine<<<NROWS / 256, 256, 0, stream>>>(Pmax, Psum, Ptv, out);
}
